// VectorVanillaTransformer_19129784336565
// MI455X (gfx1250) — matrix operand order repaired and checked by reading, not compiled, not run on hardware
//
#include <hip/hip_runtime.h>
#include <cmath>

// ---------------- problem dims (fixed by reference) ----------------
// N=512 seq, B=2, D=768, E=256 expand, F=3072 mlp, L=4 layers,
// H=32 heads, hd=24, 3hd=72, S=N+E=768.

typedef __bf16 bf16;
typedef __attribute__((ext_vector_type(16))) __bf16 v16bf;
typedef __attribute__((ext_vector_type(8)))  __bf16 v8bf;
typedef __attribute__((ext_vector_type(4)))  __bf16 v4bf;
typedef __attribute__((ext_vector_type(8)))  float  v8f;
typedef __attribute__((ext_vector_type(4)))  float  v4f;

__device__ __forceinline__ float siluf(float x) {
    // x * sigmoid(x) with v_rcp_f32 instead of the IEEE divide sequence
    return x * __builtin_amdgcn_rcpf(1.0f + __expf(-x));
}

// =====================================================================
// Generic batched tiled GEMM, bf16 WMMA, f32 accumulate.
//   C[bz] = epilogue( alpha * (opA(A[bz]) x B[bz]) + bias )
//   A: [M,K] row-major f32 (optionally silu() on load)
//   B: BT==0 -> [K,N] row-major ; BT==1 -> [N,K] row-major (i.e. B^T)
//   C: [M,N] f32
// Block: 256 threads = 8 wave32 waves, tile 128x128, K-step 32.
// Wave grid 4(M) x 2(N): each wave -> 32x64 = 2x4 WMMA 16x16 accums.
// Software pipeline: double-buffered LDS; fetch(kt+1)->regs issued
// before the 8 WMMAs of tile kt, regs committed to LDS afterwards.
// One __syncthreads per K-tile. Guarded scalar path only on edge tiles.
// =====================================================================
#define BMT 128
#define BNT 128
#define BKT 32
#define LDSK (BKT + 8)   // row stride 40 bf16 = 80B: 16B-aligned fragments

template<bool BT, bool SILU_A, bool SILU_OUT>
__global__ __launch_bounds__(256)
void gemm_wmma_kernel(const float* __restrict__ A, const float* __restrict__ Bm,
                      float* __restrict__ C,
                      int M, int N, int K,
                      long long sA, long long sB, long long sC,
                      const float* __restrict__ bias, int biasDiv, float alpha)
{
    __shared__ bf16 lsA[2][BMT * LDSK];
    __shared__ bf16 lsB[2][BNT * LDSK];

    const int t    = threadIdx.x;
    const int lane = t & 31;          // wave32
    const int wid  = t >> 5;          // 0..7
    const int wm   = wid & 3;         // 4 wave-rows
    const int wn   = wid >> 2;        // 2 wave-cols
    const int mWave = wm * 32;
    const int nWave = wn * 64;
    const int lm = lane & 15;
    const int lg = lane >> 4;

    const int bz     = blockIdx.z;
    const int blockM = blockIdx.y * BMT;
    const int blockN = blockIdx.x * BNT;

    const float* Ab = A  + (long long)bz * sA;
    const float* Bb = Bm + (long long)bz * sB;
    float*       Cb = C  + (long long)bz * sC;

    const bool aFullM = (blockM + BMT) <= M;
    const bool bFullN = (blockN + BNT) <= N;

    // tile-load thread mappings
    const int rA = t >> 3;            // 0..31 (row within 32-row slab)
    const int cA = (t & 7) * 4;       // 0..28 (k offset, 4-wide)
    const int kkB = t >> 5;           // 0..7  (normal-B: k row within 8-slab)
    const int n4B = lane * 4;         // 0..124 (normal-B: n offset, 4-wide)

    v4f ra[4];                        // staged A slab values
    v4f rb[4];                        // staged B slab values

    auto fetchA = [&](int kt) {
        const int k0 = kt * BKT;
        if (aFullM && (k0 + BKT) <= K) {
            #pragma unroll
            for (int ii = 0; ii < 4; ++ii)
                ra[ii] = *(const v4f*)(Ab + (long long)(blockM + ii * 32 + rA) * K + (k0 + cA));
        } else {
            #pragma unroll
            for (int ii = 0; ii < 4; ++ii) {
                int gm = blockM + ii * 32 + rA;
                #pragma unroll
                for (int i = 0; i < 4; ++i) {
                    int gk = k0 + cA + i;
                    ra[ii][i] = (gm < M && gk < K) ? Ab[(long long)gm * K + gk] : 0.0f;
                }
            }
        }
    };
    auto commitA = [&](int buf) {
        #pragma unroll
        for (int ii = 0; ii < 4; ++ii) {
            v4bf w;
            #pragma unroll
            for (int i = 0; i < 4; ++i) {
                float f = ra[ii][i];
                if (SILU_A) f = siluf(f);
                w[i] = (bf16)f;
            }
            *(v4bf*)&lsA[buf][(ii * 32 + rA) * LDSK + cA] = w;
        }
    };
    auto fetchB = [&](int kt) {
        const int k0 = kt * BKT;
        if (BT) {
            if (bFullN && (k0 + BKT) <= K) {
                #pragma unroll
                for (int ii = 0; ii < 4; ++ii)
                    rb[ii] = *(const v4f*)(Bb + (long long)(blockN + ii * 32 + rA) * K + (k0 + cA));
            } else {
                #pragma unroll
                for (int ii = 0; ii < 4; ++ii) {
                    int gn = blockN + ii * 32 + rA;
                    #pragma unroll
                    for (int i = 0; i < 4; ++i) {
                        int gk = k0 + cA + i;
                        rb[ii][i] = (gn < N && gk < K) ? Bb[(long long)gn * K + gk] : 0.0f;
                    }
                }
            }
        } else {
            if (bFullN && (k0 + BKT) <= K) {
                #pragma unroll
                for (int ii = 0; ii < 4; ++ii)
                    rb[ii] = *(const v4f*)(Bb + (long long)(k0 + kkB + ii * 8) * N + (blockN + n4B));
            } else {
                #pragma unroll
                for (int ii = 0; ii < 4; ++ii) {
                    int gk = k0 + kkB + ii * 8;
                    #pragma unroll
                    for (int i = 0; i < 4; ++i) {
                        int gn = blockN + n4B + i;
                        rb[ii][i] = (gk < K && gn < N) ? Bb[(long long)gk * N + gn] : 0.0f;
                    }
                }
            }
        }
    };
    auto commitB = [&](int buf) {
        if (BT) {
            #pragma unroll
            for (int ii = 0; ii < 4; ++ii) {
                v4bf w;
                #pragma unroll
                for (int i = 0; i < 4; ++i) w[i] = (bf16)rb[ii][i];
                *(v4bf*)&lsB[buf][(ii * 32 + rA) * LDSK + cA] = w;
            }
        } else {
            #pragma unroll
            for (int ii = 0; ii < 4; ++ii)
                #pragma unroll
                for (int i = 0; i < 4; ++i)
                    lsB[buf][(n4B + i) * LDSK + kkB + ii * 8] = (bf16)rb[ii][i];
        }
    };

    v8f acc[2][4];
    #pragma unroll
    for (int mi = 0; mi < 2; ++mi)
        #pragma unroll
        for (int ni = 0; ni < 4; ++ni)
            #pragma unroll
            for (int e = 0; e < 8; ++e) acc[mi][ni][e] = 0.0f;

    const int kTiles = (K + BKT - 1) / BKT;

    // preamble: stage + commit tile 0
    fetchA(0); fetchB(0);
    commitA(0); commitB(0);

    for (int kt = 0; kt < kTiles; ++kt) {
        __syncthreads();                       // tile kt visible; buf (kt+1)&1 free
        const bool more = (kt + 1) < kTiles;
        if (more) { fetchA(kt + 1); fetchB(kt + 1); }   // loads in flight during WMMA

        const int buf = kt & 1;
        // ---- fragments per ISA 7.12.2 layouts (16B-aligned LDS loads) ----
        v16bf afr[2];
        #pragma unroll
        for (int mi = 0; mi < 2; ++mi) {
            const bf16* ap = &lsA[buf][(mWave + mi * 16 + lm) * LDSK + lg * 8];
            v8bf lo = *(const v8bf*)(ap);
            v8bf hi = *(const v8bf*)(ap + 16);
            afr[mi] = __builtin_shufflevector(lo, hi, 0, 1, 2, 3, 4, 5, 6, 7,
                                              8, 9, 10, 11, 12, 13, 14, 15);
        }
        v16bf bfr[4];
        #pragma unroll
        for (int ni = 0; ni < 4; ++ni) {
            const bf16* bp = &lsB[buf][(nWave + ni * 16 + lm) * LDSK + lg * 8];
            v8bf lo = *(const v8bf*)(bp);
            v8bf hi = *(const v8bf*)(bp + 16);
            bfr[ni] = __builtin_shufflevector(lo, hi, 0, 1, 2, 3, 4, 5, 6, 7,
                                              8, 9, 10, 11, 12, 13, 14, 15);
        }
        #pragma unroll
        for (int mi = 0; mi < 2; ++mi)
            #pragma unroll
            for (int ni = 0; ni < 4; ++ni)
                acc[mi][ni] = __builtin_amdgcn_wmma_f32_16x16x32_bf16(
                    false, afr[mi], false, bfr[ni], (short)0, acc[mi][ni], false, false);

        if (more) { commitA((kt + 1) & 1); commitB((kt + 1) & 1); }
    }

    // ---- epilogue: C[lane n=lm][vgpr j -> m=lg*8+j] ----
    const float* biasB = bias ? (bias + (long long)(bz / biasDiv) * (long long)M * N) : nullptr;
    if (aFullM & bFullN) {
        #pragma unroll
        for (int mi = 0; mi < 2; ++mi) {
            #pragma unroll
            for (int ni = 0; ni < 4; ++ni) {
                #pragma unroll
                for (int j = 0; j < 8; ++j) {
                    int gm = blockM + mWave + mi * 16 + lg * 8 + j;
                    int gn = blockN + nWave + ni * 16 + lm;
                    float v = acc[mi][ni][j] * alpha;
                    if (biasB) v += biasB[(long long)gm * N + gn];
                    if (SILU_OUT) v = siluf(v);
                    Cb[(long long)gm * N + gn] = v;
                }
            }
        }
    } else {
        #pragma unroll
        for (int mi = 0; mi < 2; ++mi) {
            #pragma unroll
            for (int ni = 0; ni < 4; ++ni) {
                #pragma unroll
                for (int j = 0; j < 8; ++j) {
                    int gm = blockM + mWave + mi * 16 + lg * 8 + j;
                    int gn = blockN + nWave + ni * 16 + lm;
                    if (gm < M && gn < N) {
                        float v = acc[mi][ni][j] * alpha;
                        if (biasB) v += biasB[(long long)gm * N + gn];
                        if (SILU_OUT) v = siluf(v);
                        Cb[(long long)gm * N + gn] = v;
                    }
                }
            }
        }
    }
}

// =====================================================================
// Elementwise / reduction helper kernels
// =====================================================================
__global__ void transpose_in_kernel(const float* __restrict__ x, float* __restrict__ xb,
                                    int N, int B, int D)
{
    long long i = (long long)blockIdx.x * blockDim.x + threadIdx.x;
    long long total = (long long)N * B * D;
    if (i >= total) return;
    int d = (int)(i % D); long long r = i / D;
    int b = (int)(r % B); int n = (int)(r / B);
    xb[((long long)b * N + n) * D + d] = x[i];
}

__global__ void build_bias_kernel(const float* __restrict__ pos, const float* __restrict__ epos,
                                  const bool* __restrict__ pmask, const bool* __restrict__ emask,
                                  float* __restrict__ bias, int B, int N, int E)
{
    int S = N + E;
    long long i = (long long)blockIdx.x * blockDim.x + threadIdx.x;
    long long total = (long long)B * N * S;
    if (i >= total) return;
    int s = (int)(i % S); long long r = i / S;
    int n = (int)(r % N); int b = (int)(r / N);
    float px = pos[((long long)b * N + n) * 3 + 0];
    float py = pos[((long long)b * N + n) * 3 + 1];
    float pz = pos[((long long)b * N + n) * 3 + 2];
    float qx, qy, qz; bool kp;
    if (s < N) {
        qx = pos[((long long)b * N + s) * 3 + 0];
        qy = pos[((long long)b * N + s) * 3 + 1];
        qz = pos[((long long)b * N + s) * 3 + 2];
        kp = pmask[(long long)b * N + s];
    } else {
        int e = s - N;
        qx = epos[((long long)b * E + e) * 3 + 0];
        qy = epos[((long long)b * E + e) * 3 + 1];
        qz = epos[((long long)b * E + e) * 3 + 2];
        kp = emask[(long long)b * E + e];
    }
    float dx = px - qx, dy = py - qy, dz = pz - qz;
    float sq = dx * dx + dy * dy + dz * dz;
    float dist = sq > 0.0f ? sqrtf(sq) : 0.0f;
    if (pmask[(long long)b * N + n]) dist = 1e6f;
    if (kp) dist = 1e6f;
    dist = 1.0f / (dist + 1.0f);
    bias[i] = kp ? -1e9f : dist;
}

__global__ __launch_bounds__(256)
void ln_kernel(const float* __restrict__ in, float* __restrict__ out,
               const float* __restrict__ g, const float* __restrict__ b, int D)
{
    __shared__ float red[256];
    long long row = blockIdx.x;
    const float* x = in + row * D;
    float* y = out + row * D;
    float s = 0.0f;
    for (int d = threadIdx.x; d < D; d += 256) s += x[d];
    red[threadIdx.x] = s; __syncthreads();
    for (int o = 128; o > 0; o >>= 1) { if ((int)threadIdx.x < o) red[threadIdx.x] += red[threadIdx.x + o]; __syncthreads(); }
    float mean = red[0] / (float)D; __syncthreads();
    float v = 0.0f;
    for (int d = threadIdx.x; d < D; d += 256) { float t = x[d] - mean; v += t * t; }
    red[threadIdx.x] = v; __syncthreads();
    for (int o = 128; o > 0; o >>= 1) { if ((int)threadIdx.x < o) red[threadIdx.x] += red[threadIdx.x + o]; __syncthreads(); }
    float inv = rsqrtf(red[0] / (float)D + 1e-5f);
    for (int d = threadIdx.x; d < D; d += 256) y[d] = (x[d] - mean) * inv * g[d] + b[d];
}

__global__ __launch_bounds__(256)
void softmax_kernel(float* __restrict__ p, int S)
{
    __shared__ float red[256];
    long long row = blockIdx.x;
    float* x = p + row * S;
    float mx = -3.0e38f;
    for (int s = threadIdx.x; s < S; s += 256) mx = fmaxf(mx, x[s]);
    red[threadIdx.x] = mx; __syncthreads();
    for (int o = 128; o > 0; o >>= 1) { if ((int)threadIdx.x < o) red[threadIdx.x] = fmaxf(red[threadIdx.x], red[threadIdx.x + o]); __syncthreads(); }
    mx = red[0]; __syncthreads();
    float sum = 0.0f;
    for (int s = threadIdx.x; s < S; s += 256) { float e = __expf(x[s] - mx); x[s] = e; sum += e; }
    red[threadIdx.x] = sum; __syncthreads();
    for (int o = 128; o > 0; o >>= 1) { if ((int)threadIdx.x < o) red[threadIdx.x] += red[threadIdx.x + o]; __syncthreads(); }
    float inv = __builtin_amdgcn_rcpf(red[0]);
    for (int s = threadIdx.x; s < S; s += 256) x[s] *= inv;
}

__global__ void modulate_kernel(const float* __restrict__ vl, const float* __restrict__ scx,
                                float* __restrict__ vmod, long long total, int D)
{
    long long i = (long long)blockIdx.x * blockDim.x + threadIdx.x;
    if (i >= total) return;
    int d = (int)(i % D);
    long long r = i / D;           // (b*N+n)*3 + j
    long long bn = r / 3;
    vmod[i] = vl[i] * scx[bn * D + d];
}

__global__ void gather_kv_kernel(const float* __restrict__ vmod, const int* __restrict__ outcell,
                                 float* __restrict__ kv, int B, int N, int E, int S, int D)
{
    long long i = (long long)blockIdx.x * blockDim.x + threadIdx.x;
    long long total = (long long)B * S * 3 * D;
    if (i >= total) return;
    int d = (int)(i % D); long long r = i / D;
    int j = (int)(r % 3); long long r2 = r / 3;
    int s = (int)(r2 % S); int b = (int)(r2 / S);
    int n = (s < N) ? s : outcell[(long long)b * E + (s - N)];
    kv[i] = vmod[(((long long)b * N + n) * 3 + j) * D + d];
}

// q[b,h,n,j*hd+d] = src[b,n,j,h*hd+d]  (src seq length Ns)
__global__ void permute_qk_kernel(const float* __restrict__ src, float* __restrict__ dst,
                                  int B, int H, int Ns, int D, int hd)
{
    long long i = (long long)blockIdx.x * blockDim.x + threadIdx.x;
    int hd3 = 3 * hd;
    long long total = (long long)B * H * Ns * hd3;
    if (i >= total) return;
    int tt = (int)(i % hd3); long long r = i / hd3;
    int n = (int)(r % Ns); long long r2 = r / Ns;
    int h = (int)(r2 % H); int b = (int)(r2 / H);
    int j = tt / hd, d = tt % hd;
    dst[i] = src[(((long long)b * Ns + n) * 3 + j) * D + h * hd + d];
}

// o[b,n,j,h*hd+d] = heads[b,h,n,j*hd+d]
__global__ void permute_o_kernel(const float* __restrict__ heads, float* __restrict__ o,
                                 int B, int H, int N, int D, int hd)
{
    long long i = (long long)blockIdx.x * blockDim.x + threadIdx.x;
    long long total = (long long)B * N * 3 * D;
    if (i >= total) return;
    int dd = (int)(i % D); long long r = i / D;
    int j = (int)(r % 3); long long r2 = r / 3;
    int n = (int)(r2 % N); int b = (int)(r2 / N);
    int h = dd / hd, d = dd % hd;
    o[i] = heads[(((long long)(b * H + h) * N) + n) * (3 * hd) + j * hd + d];
}

__global__ void add_inplace_kernel(float* __restrict__ dst, const float* __restrict__ src, long long n)
{
    long long i = (long long)blockIdx.x * blockDim.x + threadIdx.x;
    if (i < n) dst[i] += src[i];
}

// x = x * sum_j sc + sum_j sh + x ; ss layout [B,N,3,2D]
__global__ void final_x_kernel(float* __restrict__ x, const float* __restrict__ ss,
                               long long total, int D)
{
    long long i = (long long)blockIdx.x * blockDim.x + threadIdx.x;
    if (i >= total) return;
    int d = (int)(i % D);
    long long bn = i / D;
    float sc = 0.0f, sh = 0.0f;
    #pragma unroll
    for (int j = 0; j < 3; ++j) {
        long long base = (bn * 3 + j) * (long long)(2 * D);
        sc += ss[base + d];
        sh += ss[base + D + d];
    }
    x[i] = x[i] * sc + sh + x[i];
}

// =====================================================================
// Host orchestration
// =====================================================================
extern "C" void kernel_launch(void* const* d_in, const int* in_sizes, int n_in,
                              void* d_out, int out_size, void* d_ws, size_t ws_size,
                              hipStream_t stream)
{
    (void)in_sizes; (void)n_in; (void)out_size; (void)ws_size;
    const float* x_in   = (const float*)d_in[0];
    const float* pos    = (const float*)d_in[1];
    const bool*  pmask  = (const bool*)d_in[2];
    const int*   outcell= (const int*)d_in[3];
    const float* epos   = (const float*)d_in[4];
    const bool*  emask  = (const bool*)d_in[5];
    const float* vproj_w      = (const float*)d_in[6];
    const float* ln_x_g       = (const float*)d_in[7];
    const float* ln_x_b       = (const float*)d_in[8];
    const float* ln_vec_g     = (const float*)d_in[9];
    const float* ln_vec_b     = (const float*)d_in[10];
    const float* adaln_x_w    = (const float*)d_in[11];
    const float* adaln_vm_w   = (const float*)d_in[12];
    const float* out_proj_w   = (const float*)d_in[13];
    const float* vmn_g        = (const float*)d_in[14];
    const float* vmn_b        = (const float*)d_in[15];
    const float* mlp_w1       = (const float*)d_in[16];
    const float* mlp_w2       = (const float*)d_in[17];

    constexpr int B = 2, N = 512, D = 768, E = 256, F = 3072, L = 4;
    constexpr int H = 32, hd = 24, HD3 = 72, S = N + E;   // S = 768
    const long long SZ_X = (long long)B * N * D;
    const long long SZ_V = (long long)B * N * 3 * D;
    const long long SZ_KV = (long long)B * S * 3 * D;
    const long long SZ_Q = (long long)B * H * N * HD3;
    const long long SZ_K = (long long)B * H * S * HD3;
    const long long SZ_L = (long long)B * H * N * S;

    float* ws = (float*)d_ws;
    long long off = 0;
    auto alloc = [&](long long nfl) { float* p = ws + off; off += (nfl + 63) & ~63LL; return p; };
    float* bias   = alloc((long long)B * N * S);
    float* xbuf   = alloc(SZ_X);
    float* vecb   = alloc(SZ_V);
    float* xl     = alloc(SZ_X);
    float* vl     = alloc(SZ_V);
    float* scx    = alloc(SZ_X);
    float* vmod   = alloc(SZ_V);
    float* kv     = alloc(SZ_KV);
    float* q      = alloc(SZ_Q);
    float* kbuf   = alloc(SZ_K);
    float* logits = alloc(SZ_L);
    float* oproj  = alloc(SZ_V);
    float* m1     = alloc((long long)B * N * 3 * F);
    float* vm2    = alloc(SZ_V);
    float* ssb    = alloc((long long)B * N * 3 * 2 * D);
    float* oheads = q;      // q free once logits computed
    float* operm  = vmod;   // vmod free once q / kv built
    float* vmln   = kv;     // kv free once k built

    auto blk1d = [](long long n) { return (unsigned)((n + 255) / 256); };

    // ------- setup: bias, x transpose, initial vec -------
    build_bias_kernel<<<blk1d((long long)B * N * S), 256, 0, stream>>>(pos, epos, pmask, emask, bias, B, N, E);
    transpose_in_kernel<<<blk1d((long long)N * B * D), 256, 0, stream>>>(x_in, xbuf, N, B, D);
    // vec = silu(x) @ vec_project_w : [B*N, 3D]
    gemm_wmma_kernel<false, true, false><<<dim3((3 * D + 127) / 128, (B * N + 127) / 128, 1), 256, 0, stream>>>(
        xbuf, vproj_w, vecb, B * N, 3 * D, D, 0, 0, 0, nullptr, 1, 1.0f);

    const float attn_scale = 1.0f / sqrtf((float)HD3);

    for (int l = 0; l < L; ++l) {
        ln_kernel<<<B * N, 256, 0, stream>>>(xbuf, xl, ln_x_g + l * D, ln_x_b + l * D, D);
        ln_kernel<<<B * N * 3, 256, 0, stream>>>(vecb, vl, ln_vec_g + l * D, ln_vec_b + l * D, D);
        // scale_x = silu(xl) @ adaln_x_w[l]
        gemm_wmma_kernel<false, true, false><<<dim3((D + 127) / 128, (B * N + 127) / 128, 1), 256, 0, stream>>>(
            xl, adaln_x_w + (long long)l * D * D, scx, B * N, D, D, 0, 0, 0, nullptr, 1, 1.0f);
        modulate_kernel<<<blk1d(SZ_V), 256, 0, stream>>>(vl, scx, vmod, SZ_V, D);
        gather_kv_kernel<<<blk1d(SZ_KV), 256, 0, stream>>>(vmod, outcell, kv, B, N, E, S, D);
        permute_qk_kernel<<<blk1d(SZ_Q), 256, 0, stream>>>(vmod, q, B, H, N, D, hd);
        permute_qk_kernel<<<blk1d(SZ_K), 256, 0, stream>>>(kv, kbuf, B, H, S, D, hd);
        // logits = q @ k^T * scale + bias  (batched over B*H; K stored [S,72] => BT mode)
        gemm_wmma_kernel<true, false, false><<<dim3((S + 127) / 128, (N + 127) / 128, B * H), 256, 0, stream>>>(
            q, kbuf, logits, N, S, HD3,
            (long long)N * HD3, (long long)S * HD3, (long long)N * S, bias, H, attn_scale);
        softmax_kernel<<<B * H * N, 256, 0, stream>>>(logits, S);
        // o = p @ v  (v == k, stored [S,72] row-major = [K,N] normal mode)
        gemm_wmma_kernel<false, false, false><<<dim3((HD3 + 127) / 128, (N + 127) / 128, B * H), 256, 0, stream>>>(
            logits, kbuf, oheads, N, HD3, S,
            (long long)N * S, (long long)S * HD3, (long long)N * HD3, nullptr, 1, 1.0f);
        permute_o_kernel<<<blk1d(SZ_V), 256, 0, stream>>>(oheads, operm, B, H, N, D, hd);
        // o @ out_proj_w[l]
        gemm_wmma_kernel<false, false, false><<<dim3((D + 127) / 128, (B * N * 3 + 127) / 128, 1), 256, 0, stream>>>(
            operm, out_proj_w + (long long)l * D * D, oproj, B * N * 3, D, D, 0, 0, 0, nullptr, 1, 1.0f);
        add_inplace_kernel<<<blk1d(SZ_X), 256, 0, stream>>>(xbuf, xl, SZ_X);     // x = xl + rx
        add_inplace_kernel<<<blk1d(SZ_V), 256, 0, stream>>>(vecb, oproj, SZ_V);  // vec = o + rv
        // MLP branch
        ln_kernel<<<B * N * 3, 256, 0, stream>>>(vecb, vmln, vmn_g + l * D, vmn_b + l * D, D);
        gemm_wmma_kernel<false, false, true><<<dim3((F + 127) / 128, (B * N * 3 + 127) / 128, 1), 256, 0, stream>>>(
            vmln, mlp_w1 + (long long)l * D * F, m1, B * N * 3, F, D, 0, 0, 0, nullptr, 1, 1.0f);  // silu(vm@w1)
        gemm_wmma_kernel<false, false, false><<<dim3((D + 127) / 128, (B * N * 3 + 127) / 128, 1), 256, 0, stream>>>(
            m1, mlp_w2 + (long long)l * F * D, vm2, B * N * 3, D, F, 0, 0, 0, nullptr, 1, 1.0f);
        gemm_wmma_kernel<false, true, false><<<dim3((2 * D + 127) / 128, (B * N * 3 + 127) / 128, 1), 256, 0, stream>>>(
            vm2, adaln_vm_w + (long long)l * D * 2 * D, ssb, B * N * 3, 2 * D, D, 0, 0, 0, nullptr, 1, 1.0f);
        final_x_kernel<<<blk1d(SZ_X), 256, 0, stream>>>(xbuf, ssb, SZ_X, D);
        add_inplace_kernel<<<blk1d(SZ_V), 256, 0, stream>>>(vecb, vm2, SZ_V);    // vec = vm + rv
    }

    hipMemcpyAsync(d_out, xbuf, sizeof(float) * SZ_X, hipMemcpyDeviceToDevice, stream);
    hipMemcpyAsync((float*)d_out + SZ_X, vecb, sizeof(float) * SZ_V, hipMemcpyDeviceToDevice, stream);
}
